// norm_conv_32538672234775
// MI455X (gfx1250) — matrix operand order repaired and checked by reading, not compiled, not run on hardware
//
#include <hip/hip_runtime.h>

typedef __attribute__((ext_vector_type(16))) _Float16 v16h;
typedef __attribute__((ext_vector_type(8)))  _Float16 v8h;
typedef __attribute__((ext_vector_type(8)))  float    v8f;

#define HW    32
#define PADW  34
#define NCH   64           // input channels
#define NOUT  64           // output channels
#define NQ    9            // 3x3 shifts
#define CSTR  72           // LDS channel stride in halves (144B: 16B aligned, bank-spread)
#define KSTR  72
#define ILDS_BYTES (PADW*PADW*CSTR*2)   // 166464
#define KLDS_BYTES (NQ*NOUT*KSTR*2)     // 82944
#define STAT_BYTES (PADW*PADW*4*2 + 1024*4*2 + NOUT*4)  // s1p,s2p,mean,rstd,S = 17696
#define SHMEM_BYTES (ILDS_BYTES + KLDS_BYTES + STAT_BYTES)  // 267104

// ---------------------------------------------------------------------------
// Prep: repack conv_kernel [576][64] f32 -> f16 [q][o][c] (padded stride) and
// column sums S[o] = sum_p W[p][o], both into workspace.
// ---------------------------------------------------------------------------
__global__ void nc_prep_kernel(const float* __restrict__ wk,
                               _Float16* __restrict__ bm,
                               float* __restrict__ S) {
    int tid = threadIdx.x;
    for (int i = tid; i < NQ * NOUT * NCH; i += 512) {
        int q = i >> 12;           // / 4096
        int r = i & 4095;
        int o = r >> 6;
        int c = r & 63;
        bm[(q * NOUT + o) * KSTR + c] = (_Float16)wk[(c * NQ + q) * NOUT + o];
    }
    if (tid < NOUT) {
        float s = 0.f;
        for (int p = 0; p < 576; ++p) s += wk[p * NOUT + tid];
        S[tid] = s;
    }
}

// ---------------------------------------------------------------------------
// Main: one block per image. Implicit-GEMM conv with folded normalization.
// ---------------------------------------------------------------------------
__global__ void __launch_bounds__(512, 1)
nc_main_kernel(const float* __restrict__ a,
               const _Float16* __restrict__ bm,
               const float* __restrict__ Sg,
               float* __restrict__ out) {
    extern __shared__ char smem[];
    _Float16* ilds  = (_Float16*)smem;                       // [34*34][72] f16
    _Float16* klds  = (_Float16*)(smem + ILDS_BYTES);        // [9][64][72] f16
    float*    s1p   = (float*)(smem + ILDS_BYTES + KLDS_BYTES);
    float*    s2p   = s1p + PADW * PADW;
    float*    meanA = s2p + PADW * PADW;
    float*    rstdA = meanA + 1024;
    float*    Ss    = rstdA + 1024;

    const int tid = threadIdx.x;
    const int b   = blockIdx.x;

    // --- phase 1: kernel tile ws -> LDS (b128 copies), S -> LDS ---
    {
        const uint4* src = (const uint4*)bm;
        uint4*       dst = (uint4*)klds;
        for (int i = tid; i < KLDS_BYTES / 16; i += 512) dst[i] = src[i];
        if (tid < NOUT) Ss[tid] = Sg[tid];
    }
    // --- phase 2: image interior, f32 global -> f16 LDS [pix][c] ---
    {
        const float4* A4 = (const float4*)(a + ((size_t)b << 16));
        for (int it = 0; it < 32; ++it) {
            int i4 = it * 512 + tid;
            float4 v = A4[i4];
            int idx = i4 << 2;
            int c   = idx >> 10;
            int rem = idx & 1023;
            int y = rem >> 5, x = rem & 31;
            int pix = (y + 1) * PADW + (x + 1);
            ilds[(pix + 0) * CSTR + c] = (_Float16)v.x;
            ilds[(pix + 1) * CSTR + c] = (_Float16)v.y;
            ilds[(pix + 2) * CSTR + c] = (_Float16)v.z;
            ilds[(pix + 3) * CSTR + c] = (_Float16)v.w;
        }
    }
    __syncthreads();
    // --- phase 2b: reflect padding, rows then columns (corners via cols) ---
    for (int i = tid; i < 2 * HW * NCH; i += 512) {
        int sel = i / (HW * NCH);
        int j   = i % (HW * NCH);
        int xp  = (j >> 6) + 1;
        int c   = j & 63;
        if (sel == 0) ilds[(0 * PADW + xp) * CSTR + c]  = ilds[(2 * PADW + xp) * CSTR + c];
        else          ilds[(33 * PADW + xp) * CSTR + c] = ilds[(31 * PADW + xp) * CSTR + c];
    }
    __syncthreads();
    for (int i = tid; i < 2 * PADW * NCH; i += 512) {
        int sel = i / (PADW * NCH);
        int j   = i % (PADW * NCH);
        int yp  = j >> 6;
        int c   = j & 63;
        if (sel == 0) ilds[(yp * PADW + 0) * CSTR + c]  = ilds[(yp * PADW + 2) * CSTR + c];
        else          ilds[(yp * PADW + 33) * CSTR + c] = ilds[(yp * PADW + 31) * CSTR + c];
    }
    __syncthreads();
    // --- phase 3: per-pixel channel sums (for patch mean/std) ---
    for (int p = tid; p < PADW * PADW; p += 512) {
        const _Float16* row = ilds + p * CSTR;
        float s1 = 0.f, s2 = 0.f;
#pragma unroll 8
        for (int c = 0; c < NCH; ++c) { float v = (float)row[c]; s1 += v; s2 += v * v; }
        s1p[p] = s1; s2p[p] = s2;
    }
    __syncthreads();
    // --- phase 4: per-patch mean / unbiased rstd via 3x3 box of sums ---
    for (int l = tid; l < 1024; l += 512) {
        int y = l >> 5, x = l & 31;
        float s = 0.f, ss = 0.f;
#pragma unroll
        for (int kh = 0; kh < 3; ++kh)
#pragma unroll
            for (int kw = 0; kw < 3; ++kw) {
                int p = (y + kh) * PADW + (x + kw);
                s += s1p[p]; ss += s2p[p];
            }
        float mean = s * (1.f / 576.f);
        float var  = (ss - s * s * (1.f / 576.f)) * (1.f / 575.f);
        meanA[l] = mean;
        rstdA[l] = rsqrtf(fmaxf(var, 1e-20f));
    }
    __syncthreads();

    // --- phase 5: WMMA GEMM.  M=64 outputs, N=1024 pixels, K = 9 shifts x 64ch ---
    const int w     = tid >> 5;          // wave 0..15 -> pixel chunk w*64
    const int lane  = tid & 31;
    const int lrow  = lane & 15;
    const int half8 = (lane >> 4) << 3;  // A-frag / D-frag upper-half offset
    const int half16 = (lane >> 4) << 4; // unused (B frags use half8)

    int pixOff[4];
#pragma unroll
    for (int nt = 0; nt < 4; ++nt) {
        int l = w * 64 + nt * 16 + lrow;
        int y = l >> 5, x = l & 31;
        pixOff[nt] = y * PADW + x;       // patch top-left in padded coords
    }

    v8f acc[4][4];
#pragma unroll
    for (int mt = 0; mt < 4; ++mt)
#pragma unroll
        for (int nt = 0; nt < 4; ++nt) acc[mt][nt] = {};

    for (int q = 0; q < NQ; ++q) {
        int kh = q / 3, kw = q % 3;
        int sh = kh * PADW + kw;
#pragma unroll
        for (int cb = 0; cb < 2; ++cb) {
            const int c0 = cb * 32;
            v16h af[4], bf[4];
            // A fragments (kernel): 16-bit A layout, interleaved K (+0..7, +16..23)
#pragma unroll
            for (int mt = 0; mt < 4; ++mt) {
                const _Float16* p = klds + (q * NOUT + mt * 16 + lrow) * KSTR + c0 + half8;
                v8h lo = *(const v8h*)p;
                v8h hi = *(const v8h*)(p + 16);
                af[mt] = __builtin_shufflevector(lo, hi, 0,1,2,3,4,5,6,7,8,9,10,11,12,13,14,15);
            }
            // B fragments (patches): 16-bit B layout, interleaved K (+0..7, +16..23)
#pragma unroll
            for (int nt = 0; nt < 4; ++nt) {
                const _Float16* p = ilds + (pixOff[nt] + sh) * CSTR + c0 + half8;
                v8h lo = *(const v8h*)p;
                v8h hi = *(const v8h*)(p + 16);
                bf[nt] = __builtin_shufflevector(lo, hi, 0,1,2,3,4,5,6,7,8,9,10,11,12,13,14,15);
            }
#pragma unroll
            for (int mt = 0; mt < 4; ++mt)
#pragma unroll
                for (int nt = 0; nt < 4; ++nt)
                    acc[mt][nt] = __builtin_amdgcn_wmma_f32_16x16x32_f16(
                        false, af[mt], false, bf[nt], (short)0, acc[mt][nt], false, false);
        }
    }

    // --- epilogue: out = (acc - mean*S[o]) * rstd, coalesced stores ---
    float meanL[4], rstdL[4];
#pragma unroll
    for (int nt = 0; nt < 4; ++nt) {
        int l = w * 64 + nt * 16 + lrow;
        meanL[nt] = meanA[l];
        rstdL[nt] = rstdA[l];
    }
    float* outB = out + ((size_t)b << 16);
#pragma unroll
    for (int mt = 0; mt < 4; ++mt) {
#pragma unroll
        for (int r = 0; r < 8; ++r) {
            int o = mt * 16 + r + half8;         // D: M = vgpr (+8 upper lanes)
            float so = Ss[o];
            float* orow = outB + (o << 10) + w * 64 + lrow;
#pragma unroll
            for (int nt = 0; nt < 4; ++nt) {
                float v = (acc[mt][nt][r] - meanL[nt] * so) * rstdL[nt];
                orow[nt * 16] = v;               // D: N = lane%16 -> contiguous pixels
            }
        }
    }
}

extern "C" void kernel_launch(void* const* d_in, const int* in_sizes, int n_in,
                              void* d_out, int out_size, void* d_ws, size_t ws_size,
                              hipStream_t stream) {
    (void)in_sizes; (void)n_in; (void)out_size; (void)ws_size;
    const float* a  = (const float*)d_in[0];          // [256,64,32,32] f32
    const float* wk = (const float*)d_in[1];          // [576,64] f32
    float* out = (float*)d_out;                       // [256,64,32,32] f32
    _Float16* bm = (_Float16*)d_ws;                   // repacked f16 kernel
    float* S = (float*)((char*)d_ws + KLDS_BYTES);    // column sums

    hipFuncSetAttribute(reinterpret_cast<const void*>(nc_main_kernel),
                        hipFuncAttributeMaxDynamicSharedMemorySize, SHMEM_BYTES);

    nc_prep_kernel<<<1, 512, 0, stream>>>(wk, bm, S);
    nc_main_kernel<<<256, 512, SHMEM_BYTES, stream>>>(a, bm, S, out);
}
